// Pointnet_SA_MSG_58317065945845
// MI455X (gfx1250) — matrix operand order repaired and checked by reading, not compiled, not run on hardware
//
#include <hip/hip_runtime.h>
#include <hip/hip_bf16.h>

#define B_   8
#define N_   4096
#define S_   1024
#define CIN_ 64
#define KMAX 64

typedef __attribute__((ext_vector_type(16))) __bf16 v16bf;
typedef __attribute__((ext_vector_type(8)))  __bf16 v8bf;
typedef __attribute__((ext_vector_type(8)))  float  v8f;

__device__ __forceinline__ unsigned short f2bf(float x) {
    unsigned int u = __float_as_uint(x);
    u += 0x7FFFu + ((u >> 16) & 1u);   // round-to-nearest-even
    return (unsigned short)(u >> 16);
}

// ---------------------------------------------------------------------------
// Kernel 1: farthest point sampling. 1 block per batch, 1024 threads,
// 4 points per thread held in registers; new_xyz written straight to d_out.
// ---------------------------------------------------------------------------
__global__ void __launch_bounds__(1024)
fps_kernel(const float* __restrict__ xyz, float* __restrict__ new_xyz) {
    const int b = blockIdx.x;
    const int t = threadIdx.x;
    const float* X = xyz + (size_t)b * N_ * 3;

    float px[4], py[4], pz[4], dmin[4];
#pragma unroll
    for (int i = 0; i < 4; ++i) {
        int p = t + i * 1024;
        px[i] = X[p * 3 + 0]; py[i] = X[p * 3 + 1]; pz[i] = X[p * 3 + 2];
        dmin[i] = 1e10f;
    }

    __shared__ float s_red[32];
    __shared__ int   s_idx[32];
    __shared__ float s_c[3];
    __shared__ int   s_far;
    if (t == 0) s_far = 0;
    __syncthreads();

    const int lane = t & 31, wv = t >> 5;
    for (int it = 0; it < S_; ++it) {
        if (t == 0) {
            int f = s_far;
            float cx = X[f * 3 + 0], cy = X[f * 3 + 1], cz = X[f * 3 + 2];
            s_c[0] = cx; s_c[1] = cy; s_c[2] = cz;
            float* o = new_xyz + ((size_t)b * S_ + it) * 3;
            o[0] = cx; o[1] = cy; o[2] = cz;
        }
        __syncthreads();
        float cx = s_c[0], cy = s_c[1], cz = s_c[2];
        float best = -1.f; int bi = 0;
#pragma unroll
        for (int i = 0; i < 4; ++i) {
            float dx = px[i] - cx, dy = py[i] - cy, dz = pz[i] - cz;
            float d = dx * dx + dy * dy + dz * dz;
            dmin[i] = fminf(dmin[i], d);
            int p = t + i * 1024;
            if (dmin[i] > best || (dmin[i] == best && p < bi)) { best = dmin[i]; bi = p; }
        }
#pragma unroll
        for (int off = 16; off > 0; off >>= 1) {
            float ob = __shfl_xor(best, off, 32);
            int   oi = __shfl_xor(bi,   off, 32);
            if (ob > best || (ob == best && oi < bi)) { best = ob; bi = oi; }
        }
        if (lane == 0) { s_red[wv] = best; s_idx[wv] = bi; }
        __syncthreads();
        if (wv == 0) {
            best = s_red[lane]; bi = s_idx[lane];
#pragma unroll
            for (int off = 16; off > 0; off >>= 1) {
                float ob = __shfl_xor(best, off, 32);
                int   oi = __shfl_xor(bi,   off, 32);
                if (ob > best || (ob == best && oi < bi)) { best = ob; bi = oi; }
            }
            if (lane == 0) s_far = bi;
        }
        __syncthreads();
    }
}

// ---------------------------------------------------------------------------
// Kernel 2: top-64 kNN per query (ascending distance -> branch K uses prefix).
// 1 block per (b,s) query, 256 threads, 16 distances/thread in registers.
// ---------------------------------------------------------------------------
__global__ void __launch_bounds__(256)
knn_kernel(const float* __restrict__ xyz, const float* __restrict__ new_xyz,
           int* __restrict__ knn) {
    const int q = blockIdx.x;
    const int b = q >> 10;            // / S_
    const int t = threadIdx.x;
    const float* X = xyz + (size_t)b * N_ * 3;
    const float* Q = new_xyz + (size_t)q * 3;
    const float qx = Q[0], qy = Q[1], qz = Q[2];

    float d[16];
#pragma unroll
    for (int j = 0; j < 16; ++j) {
        int p = t + j * 256;
        float dx = X[p * 3 + 0] - qx, dy = X[p * 3 + 1] - qy, dz = X[p * 3 + 2] - qz;
        d[j] = dx * dx + dy * dy + dz * dz;
    }

    __shared__ float s_v[8];
    __shared__ int   s_i[8];
    __shared__ int   s_w;
    const int lane = t & 31, wv = t >> 5;
    int* out = knn + (size_t)q * KMAX;

    for (int r = 0; r < KMAX; ++r) {
        float best = 3.0e38f; int bi = 0x7fffffff;
#pragma unroll
        for (int j = 0; j < 16; ++j) {
            int p = t + j * 256;
            if (d[j] < best || (d[j] == best && p < bi)) { best = d[j]; bi = p; }
        }
#pragma unroll
        for (int off = 16; off > 0; off >>= 1) {
            float ob = __shfl_xor(best, off, 32);
            int   oi = __shfl_xor(bi,   off, 32);
            if (ob < best || (ob == best && oi < bi)) { best = ob; bi = oi; }
        }
        if (lane == 0) { s_v[wv] = best; s_i[wv] = bi; }
        __syncthreads();
        if (t == 0) {
            float bb = s_v[0]; int ii = s_i[0];
            for (int w = 1; w < 8; ++w)
                if (s_v[w] < bb || (s_v[w] == bb && s_i[w] < ii)) { bb = s_v[w]; ii = s_i[w]; }
            s_w = ii;
            out[r] = ii;
        }
        __syncthreads();
        int w = s_w;
        if ((w & 255) == t) d[w >> 8] = 3.4e38f;   // owner retires winner
        __syncthreads();
    }
}

// ---------------------------------------------------------------------------
// WMMA fragment loaders (bf16, 16x16x32 shapes, wave32 layouts per ISA 7.12.2)
// ---------------------------------------------------------------------------
__device__ __forceinline__ v16bf load_a_frag(const unsigned short* in, int ld,
                                             int m0, int kb, int half, int r) {
    // A 16x32: lanes 0-15 hold K 0-7 / 16-23, lanes 16-31 hold K 8-15 / 24-31
    const unsigned short* base = in + (m0 + r) * ld + kb + half * 8;
    v8bf lo = *(const v8bf*)(base);        // K = kb + half*8 + [0..7]
    v8bf hi = *(const v8bf*)(base + 16);   // K = kb + 16 + half*8 + [0..7]
    return __builtin_shufflevector(lo, hi, 0,1,2,3,4,5,6,7,8,9,10,11,12,13,14,15);
}

__device__ __forceinline__ v16bf load_b_frag(const unsigned short* wt, int ld,
                                             int n0, int kb, int half, int r) {
    // B 32x16: lane = column; lanes 0-15 K 0-7 / 16-23, lanes 16-31 K 8-15 / 24-31.
    // Weights stored transposed WT[cout][cin] so 16 K-values are contiguous.
    const unsigned short* base = wt + (n0 + r) * ld + kb + half * 8;
    v8bf lo = *(const v8bf*)(base);
    v8bf hi = *(const v8bf*)(base + 16);
    return __builtin_shufflevector(lo, hi, 0,1,2,3,4,5,6,7,8,9,10,11,12,13,14,15);
}

// One hidden MLP layer: out = relu(in @ W + b), bf16 in/out, f32 accumulate.
// All dims compile-time: full unroll + immediate addressing around the WMMAs.
template <int M, int N, int Kd, int LDIN, int LDO>
__device__ __forceinline__ void mlp_layer(const unsigned short* in,
                                          const unsigned short* wt,
                                          const float* bias,
                                          unsigned short* out) {
    const int tid  = threadIdx.x;
    const int lane = tid & 31, wave = tid >> 5;
    const int half = lane >> 4, r = lane & 15;
    constexpr int MT = M >> 4, NT = N >> 4, KT = Kd >> 5;
    for (int tix = wave; tix < MT * NT; tix += 8) {     // wave-uniform
        int mt = tix / NT, nt = tix % NT;
        v8f acc = {0.f, 0.f, 0.f, 0.f, 0.f, 0.f, 0.f, 0.f};
#pragma unroll
        for (int kc = 0; kc < KT; ++kc) {
            v16bf a = load_a_frag(in, LDIN, mt * 16, kc * 32, half, r);
            v16bf w = load_b_frag(wt, Kd, nt * 16, kc * 32, half, r);
            acc = __builtin_amdgcn_wmma_f32_16x16x32_bf16(
                false, a, false, w, (short)0, acc, false, false);
        }
        int n = nt * 16 + r;
        float bv = bias[n];
#pragma unroll
        for (int i = 0; i < 8; ++i) {    // D layout: lane holds col n, rows half*8+i
            float v = acc[i] + bv;
            v = v > 0.f ? v : 0.f;
            out[(mt * 16 + half * 8 + i) * LDO + n] = f2bf(v);
        }
    }
}

// Final layer: relu(in @ W + b) then max over the K rows, straight from the
// accumulator via integer atomicMax on float bits (valid: relu output >= 0).
template <int M, int N, int Kd, int LDIN>
__device__ __forceinline__ void mlp_final(const unsigned short* in,
                                          const unsigned short* wt,
                                          const float* bias, int* outb) {
    const int tid  = threadIdx.x;
    const int lane = tid & 31, wave = tid >> 5;
    const int half = lane >> 4, r = lane & 15;
    constexpr int MT = M >> 4, NT = N >> 4, KT = Kd >> 5;
    for (int tix = wave; tix < MT * NT; tix += 8) {
        int mt = tix / NT, nt = tix % NT;
        v8f acc = {0.f, 0.f, 0.f, 0.f, 0.f, 0.f, 0.f, 0.f};
#pragma unroll
        for (int kc = 0; kc < KT; ++kc) {
            v16bf a = load_a_frag(in, LDIN, mt * 16, kc * 32, half, r);
            v16bf w = load_b_frag(wt, Kd, nt * 16, kc * 32, half, r);
            acc = __builtin_amdgcn_wmma_f32_16x16x32_bf16(
                false, a, false, w, (short)0, acc, false, false);
        }
        int n = nt * 16 + r;
        float bv = bias[n];
        float mx = 0.f;
#pragma unroll
        for (int i = 0; i < 8; ++i) {
            float v = acc[i] + bv;
            v = v > 0.f ? v : 0.f;
            mx = fmaxf(mx, v);
        }
        atomicMax(&outb[n], __float_as_int(mx));
    }
}

// ---------------------------------------------------------------------------
// Kernel 3: one branch = gather(K,67) -> 3x (WMMA matmul + bias + relu) ->
// maxpool over K. 1 block per query, 8 waves, dynamic LDS.
// ---------------------------------------------------------------------------
template <int Ks, int C1, int C2, int C3>
__global__ void __launch_bounds__(256)
branch_kernel(const float* __restrict__ xyz, const float* __restrict__ points,
              const float* __restrict__ new_xyz, const int* __restrict__ knn,
              const float* __restrict__ W1, const float* __restrict__ bb1,
              const float* __restrict__ W2, const float* __restrict__ bb2,
              const float* __restrict__ W3, const float* __restrict__ bb3,
              float* __restrict__ outp, int out_off) {
    extern __shared__ char smem[];
    const int q = blockIdx.x;
    const int b = q >> 10;
    const int tid = threadIdx.x;

    unsigned short* wt1 = (unsigned short*)smem;       // C1 x 96 (cin padded 67->96)
    unsigned short* wt2 = wt1 + C1 * 96;               // C2 x C1
    unsigned short* wt3 = wt2 + C2 * C1;               // C3 x C2
    unsigned short* fA  = wt3 + C3 * C2;               // Ks x 96
    unsigned short* fB  = fA + Ks * 96;                // Ks x 96
    float* bias1 = (float*)(fB + Ks * 96);
    float* bias2 = bias1 + C1;
    float* bias3 = bias2 + C2;
    int*   outb  = (int*)(bias3 + C3);

    // stage transposed bf16 weights + biases into LDS
    for (int e = tid; e < C1 * 96; e += 256) {
        int co = e / 96, ci = e % 96;
        wt1[e] = (ci < CIN_ + 3) ? f2bf(W1[ci * C1 + co]) : (unsigned short)0;
    }
    for (int e = tid; e < C2 * C1; e += 256) {
        int co = e / C1, ci = e % C1;
        wt2[e] = f2bf(W2[ci * C2 + co]);
    }
    for (int e = tid; e < C3 * C2; e += 256) {
        int co = e / C2, ci = e % C2;
        wt3[e] = f2bf(W3[ci * C3 + co]);
    }
    for (int e = tid; e < C1; e += 256) bias1[e] = bb1[e];
    for (int e = tid; e < C2; e += 256) bias2[e] = bb2[e];
    for (int e = tid; e < C3; e += 256) { bias3[e] = bb3[e]; outb[e] = 0; }

    // gather grouped features: [points(64) | xyz - center (3) | zero pad] -> bf16
    const int* nn = knn + (size_t)q * KMAX;   // ascending distance; use first Ks
    const float* Q  = new_xyz + (size_t)q * 3;
    const float cx = Q[0], cy = Q[1], cz = Q[2];
    const float* Xb = xyz + (size_t)b * N_ * 3;
    const float* Pb = points + (size_t)b * N_ * CIN_;
    for (int e = tid; e < Ks * 96; e += 256) {
        int k = e / 96, c = e % 96;
        int p = nn[k];
        float v;
        if (c < CIN_)      v = Pb[(size_t)p * CIN_ + c];
        else if (c == 64)  v = Xb[p * 3 + 0] - cx;
        else if (c == 65)  v = Xb[p * 3 + 1] - cy;
        else if (c == 66)  v = Xb[p * 3 + 2] - cz;
        else               v = 0.f;
        fA[e] = f2bf(v);
    }
    __syncthreads();

    mlp_layer<Ks, C1, 96, 96, C1>(fA, wt1, bias1, fB);
    __syncthreads();
    mlp_layer<Ks, C2, C1, C1, C2>(fB, wt2, bias2, fA);
    __syncthreads();
    mlp_final<Ks, C3, C2, C2>(fA, wt3, bias3, outb);
    __syncthreads();

    float* o = outp + (size_t)q * 320 + out_off;
    for (int c = tid; c < C3; c += 256) o[c] = __int_as_float(outb[c]);
}

// ---------------------------------------------------------------------------
template <int Ks, int C1, int C2, int C3>
static void launch_branch(const float* xyz, const float* points,
                          const float* new_xyz, const int* knn,
                          void* const* d_in, int br, float* feat, int off,
                          hipStream_t stream) {
    const float* W1 = (const float*)d_in[2 + br * 6 + 0];
    const float* b1 = (const float*)d_in[2 + br * 6 + 1];
    const float* W2 = (const float*)d_in[2 + br * 6 + 2];
    const float* b2 = (const float*)d_in[2 + br * 6 + 3];
    const float* W3 = (const float*)d_in[2 + br * 6 + 4];
    const float* b3 = (const float*)d_in[2 + br * 6 + 5];
    constexpr size_t sh = 2u * ((size_t)C1 * 96 + (size_t)C2 * C1 + (size_t)C3 * C2
                                + 2u * (size_t)Ks * 96)
                        + 4u * ((size_t)C1 + C2 + C3) + 4u * (size_t)C3;
    branch_kernel<Ks, C1, C2, C3><<<B_ * S_, 256, sh, stream>>>(
        xyz, points, new_xyz, knn, W1, b1, W2, b2, W3, b3, feat, off);
}

extern "C" void kernel_launch(void* const* d_in, const int* in_sizes, int n_in,
                              void* d_out, int out_size, void* d_ws, size_t ws_size,
                              hipStream_t stream) {
    const float* xyz    = (const float*)d_in[0];
    const float* points = (const float*)d_in[1];
    float* out     = (float*)d_out;
    float* new_xyz = out;                         // (B,S,3)
    float* feat    = out + (size_t)B_ * S_ * 3;   // (B,S,320)
    int*   knn     = (int*)d_ws;                  // (B,S,64)

    fps_kernel<<<B_, 1024, 0, stream>>>(xyz, new_xyz);
    knn_kernel<<<B_ * S_, 256, 0, stream>>>(xyz, new_xyz, knn);

    launch_branch<16, 32, 32,  64>(xyz, points, new_xyz, knn, d_in, 0, feat,   0, stream);
    launch_branch<32, 64, 64, 128>(xyz, points, new_xyz, knn, d_in, 1, feat,  64, stream);
    launch_branch<64, 64, 96, 128>(xyz, points, new_xyz, knn, d_in, 2, feat, 192, stream);
}
